// FluxGAT_32238024523924
// MI455X (gfx1250) — compile-verified, matrix operand order checked by reading, not run on hardware
//
#include <hip/hip_runtime.h>
#include <hip/hip_bf16.h>

typedef __attribute__((ext_vector_type(8)))  _Float16 half8;
typedef __attribute__((ext_vector_type(16))) _Float16 v16h;
typedef __attribute__((ext_vector_type(8)))  float    v8f;

#define NTILES 19   // ceil(300/16)
#define HC 300
#define Cc 150
#define FIN 256
#define Dd 150

// ---------- helpers ----------
__device__ __forceinline__ float wred_sum(float v) {
    for (int o = 16; o; o >>= 1) v += __shfl_xor(v, o, 32);
    return v;
}
__device__ __forceinline__ unsigned fmap(float f) {
    unsigned u = __float_as_uint(f);
    return (u & 0x80000000u) ? ~u : (u | 0x80000000u);
}
__device__ __forceinline__ float finv(unsigned u) {
    unsigned b = (u & 0x80000000u) ? (u & 0x7FFFFFFFu) : ~u;
    return __uint_as_float(b);
}
__device__ __forceinline__ float lrelu(float v, float s) { return v > 0.f ? v : s * v; }

// ---------- fills ----------
__global__ void k_fill_u32(unsigned* p, unsigned v, size_t n) {
    size_t i = (size_t)blockIdx.x * blockDim.x + threadIdx.x;
    if (i < n) p[i] = v;
}
__global__ void k_fill_f32(float* p, float v, size_t n) {
    size_t i = (size_t)blockIdx.x * blockDim.x + threadIdx.x;
    if (i < n) p[i] = v;
}

// ---------- pack W into WMMA B-fragment order ----------
// packed[((nt*KT + kt)*32 + lane)*16 + j]; lane<16: K {0..7,16..23}, lane>=16: K {8..15,24..31}
__global__ void k_packW(const float* __restrict__ W, _Float16* __restrict__ out,
                        int Kreal, int Nreal, int KT, int NT) {
    int t = blockIdx.x * blockDim.x + threadIdx.x;
    int total = NT * KT * 32;
    if (t >= total) return;
    int lane = t & 31;
    int kt = (t >> 5) % KT;
    int nt = (t >> 5) / KT;
    int hi = (lane >> 4) & 1;
    int n  = nt * 16 + (lane & 15);
    _Float16* dst = out + (size_t)t * 16;
#pragma unroll
    for (int j = 0; j < 16; ++j) {
        int kk = kt * 32 + ((j < 8) ? (hi * 8 + j) : (16 + hi * 8 + (j - 8)));
        float v = (kk < Kreal && n < Nreal) ? W[(size_t)kk * Nreal + n] : 0.f;
        dst[j] = (_Float16)v;
    }
}

// ---------- argmax(x) -> gather emb -> f16 padded ----------
__global__ void k_argmax_embed(const float* __restrict__ x, const float* __restrict__ emb,
                               _Float16* __restrict__ h0, int N) {
    int wid = (int)(((size_t)blockIdx.x * blockDim.x + threadIdx.x) >> 5);
    int lane = threadIdx.x & 31;
    if (wid >= N) return;
    const float* row = x + (size_t)wid * FIN;
    float best = -3.4e38f; int bi = 0;
#pragma unroll
    for (int j = 0; j < FIN / 32; ++j) {
        int c = lane + 32 * j;
        float v = row[c];
        if (v > best || (v == best && c < bi)) { best = v; bi = c; }
    }
    for (int o = 16; o; o >>= 1) {
        float ov = __shfl_xor(best, o, 32);
        int   oi = __shfl_xor(bi, o, 32);
        if (ov > best || (ov == best && oi < bi)) { best = ov; bi = oi; }
    }
    const float* e = emb + (size_t)bi * Dd;
    _Float16* hr = h0 + (size_t)wid * 160;
    for (int c = lane; c < 160; c += 32)
        hr[c] = (c < Dd) ? (_Float16)e[c] : (_Float16)0.f;
}

// ---------- WMMA GEMM: C[N,300] = A[N,Kpad](f16) * packedB ----------
__global__ __launch_bounds__(256) void k_gemm(const _Float16* __restrict__ A, int pitchA, int KT,
                                              const _Float16* __restrict__ Bp,
                                              float* __restrict__ C, int Nrows, int Ncols) {
    __shared__ _Float16 As[16 * 320];
    int R = blockIdx.x * 16;
    int tid = threadIdx.x;
    int words = 2 * pitchA;                 // uint4 count: 16*pitchA halves
    const uint4* src = (const uint4*)(A + (size_t)R * pitchA);
    uint4* dstv = (uint4*)As;
    for (int i = tid; i < words; i += 256) {
        int rowInTile = (i * 8) / pitchA;
        uint4 z; z.x = z.y = z.z = z.w = 0u;
        dstv[i] = (R + rowInTile < Nrows) ? src[i] : z;
    }
    __syncthreads();
    int w = tid >> 5, lane = tid & 31;
    int m = lane & 15, hi = lane >> 4;
    for (int nt = w; nt < NTILES; nt += 8) {
        v8f acc = {};
        for (int kt = 0; kt < KT; ++kt) {
            const half8 a0 = *(const half8*)(As + m * pitchA + kt * 32 + hi * 8);
            const half8 a1 = *(const half8*)(As + m * pitchA + kt * 32 + 16 + hi * 8);
            v16h av = __builtin_shufflevector(a0, a1, 0, 1, 2, 3, 4, 5, 6, 7,
                                              8, 9, 10, 11, 12, 13, 14, 15);
            const _Float16* bbase = Bp + ((size_t)(nt * KT + kt) * 32 + lane) * 16;
            if (kt + 1 < KT)
                __builtin_prefetch((const void*)(bbase + 512), 0, 0); // next kt fragment
            const half8 b0 = *(const half8*)bbase;
            const half8 b1 = *(const half8*)(bbase + 8);
            v16h bv = __builtin_shufflevector(b0, b1, 0, 1, 2, 3, 4, 5, 6, 7,
                                              8, 9, 10, 11, 12, 13, 14, 15);
            acc = __builtin_amdgcn_wmma_f32_16x16x32_f16(false, av, false, bv,
                                                         (short)0, acc, false, false);
        }
        int col = nt * 16 + m;
        if (col < Ncols) {
#pragma unroll
            for (int r = 0; r < 8; ++r) {
                int row = R + r + hi * 8;
                if (row < Nrows) C[(size_t)row * Ncols + col] = acc[r];
            }
        }
    }
}

// ---------- attention terms: al_s/al_d [N,2] ----------
__global__ void k_attn(const float* __restrict__ hp, const float* __restrict__ a_s,
                       const float* __restrict__ a_d, float* __restrict__ als,
                       float* __restrict__ ald, int N) {
    int wid = (int)(((size_t)blockIdx.x * blockDim.x + threadIdx.x) >> 5);
    int lane = threadIdx.x & 31;
    if (wid >= N) return;
    const float* hr = hp + (size_t)wid * HC;
    float s0 = 0.f, s1 = 0.f, d0 = 0.f, d1 = 0.f;
    for (int c = lane; c < HC; c += 32) {
        float v = hr[c], vs = a_s[c], vd = a_d[c];
        if (c < Cc) { s0 += v * vs; d0 += v * vd; }
        else        { s1 += v * vs; d1 += v * vd; }
    }
    s0 = wred_sum(s0); s1 = wred_sum(s1); d0 = wred_sum(d0); d1 = wred_sum(d1);
    if (lane == 0) {
        als[wid * 2 + 0] = s0; als[wid * 2 + 1] = s1;
        ald[wid * 2 + 0] = d0; ald[wid * 2 + 1] = d1;
    }
}

// ---------- edge passes ----------
__global__ void k_edge_max(const int* __restrict__ ei, int E, int N,
                           const float* __restrict__ als, const float* __restrict__ ald,
                           unsigned* __restrict__ m) {
    int e = (int)((size_t)blockIdx.x * blockDim.x + threadIdx.x);
    int Etot = E + N;
    if (e >= Etot) return;
    int s, d;
    if (e < E) { s = ei[e]; d = ei[E + e]; } else { s = d = e - E; }
#pragma unroll
    for (int h = 0; h < 2; ++h) {
        float v = lrelu(als[s * 2 + h] + ald[d * 2 + h], 0.2f);
        atomicMax(&m[d * 2 + h], fmap(v));
    }
}

__global__ void k_edge_expsum(const int* __restrict__ ei, int E, int N,
                              const float* __restrict__ als, const float* __restrict__ ald,
                              const unsigned* __restrict__ m, float* __restrict__ den) {
    int e = (int)((size_t)blockIdx.x * blockDim.x + threadIdx.x);
    int Etot = E + N;
    if (e >= Etot) return;
    int s, d;
    if (e < E) { s = ei[e]; d = ei[E + e]; } else { s = d = e - E; }
#pragma unroll
    for (int h = 0; h < 2; ++h) {
        float v = lrelu(als[s * 2 + h] + ald[d * 2 + h], 0.2f);
        float ex = __expf(v - finv(m[d * 2 + h]));
        atomicAdd(&den[d * 2 + h], ex);
    }
}

__global__ void k_edge_agg(const int* __restrict__ ei, int E, int N,
                           const float* __restrict__ als, const float* __restrict__ ald,
                           const unsigned* __restrict__ m, const float* __restrict__ den,
                           const float* __restrict__ hp, float* __restrict__ agg) {
    int wid = (int)(((size_t)blockIdx.x * blockDim.x + threadIdx.x) >> 5);
    int lane = threadIdx.x & 31;
    int Etot = E + N;
    if (wid >= Etot) return;
    int s, d;
    if (wid < E) { s = ei[wid]; d = ei[E + wid]; } else { s = d = wid - E; }
    float alpha[2];
#pragma unroll
    for (int h = 0; h < 2; ++h) {
        float v = lrelu(als[s * 2 + h] + ald[d * 2 + h], 0.2f);
        float ex = __expf(v - finv(m[d * 2 + h]));
        alpha[h] = ex / den[d * 2 + h];
    }
    const float* hs = hp + (size_t)s * HC;
    float* ar = agg + (size_t)d * HC;
    for (int c = lane; c < HC; c += 32) {
        float a = (c < Cc) ? alpha[0] : alpha[1];
        atomicAdd(&ar[c], a * hs[c]);
    }
}

// ---------- +bias, LayerNorm, leaky(0.01), -> f16 padded(320) ----------
__global__ void k_ln(const float* __restrict__ agg, const float* __restrict__ b1,
                     const float* __restrict__ g, const float* __restrict__ bb,
                     _Float16* __restrict__ h1, int N) {
    int wid = (int)(((size_t)blockIdx.x * blockDim.x + threadIdx.x) >> 5);
    int lane = threadIdx.x & 31;
    if (wid >= N) return;
    const float* ar = agg + (size_t)wid * HC;
    float s = 0.f, ss = 0.f;
    for (int c = lane; c < HC; c += 32) {
        float v = ar[c] + b1[c];
        s += v; ss += v * v;
    }
    s = wred_sum(s); ss = wred_sum(ss);
    float mu = s / (float)HC;
    float var = ss / (float)HC - mu * mu;
    float rs = rsqrtf(var + 1e-5f);
    _Float16* hr = h1 + (size_t)wid * 320;
    for (int c = lane; c < 320; c += 32) {
        if (c < HC) {
            float v = ar[c] + b1[c];
            float y = (v - mu) * rs * g[c] + bb[c];
            hr[c] = (_Float16)lrelu(y, 0.01f);
        } else hr[c] = (_Float16)0.f;
    }
}

// ---------- final: leaky(agg2+b2,0.01) @ fc_w + fc_b ----------
__global__ void k_final(const float* __restrict__ agg, const float* __restrict__ b2,
                        const float* __restrict__ fcw, const float* __restrict__ fcb,
                        float* __restrict__ out, int N) {
    int wid = (int)(((size_t)blockIdx.x * blockDim.x + threadIdx.x) >> 5);
    int lane = threadIdx.x & 31;
    if (wid >= N) return;
    const float* ar = agg + (size_t)wid * HC;
    float acc = 0.f;
    for (int c = lane; c < HC; c += 32)
        acc += lrelu(ar[c] + b2[c], 0.01f) * fcw[c];
    acc = wred_sum(acc);
    if (lane == 0) out[wid] = acc + fcb[0];
}

extern "C" void kernel_launch(void* const* d_in, const int* in_sizes, int n_in,
                              void* d_out, int out_size, void* d_ws, size_t ws_size,
                              hipStream_t stream) {
    const float* x   = (const float*)d_in[0];
    const int*   ei  = (const int*)d_in[1];
    const float* emb = (const float*)d_in[3];
    const float* W1  = (const float*)d_in[4];
    const float* as1 = (const float*)d_in[5];
    const float* ad1 = (const float*)d_in[6];
    const float* b1  = (const float*)d_in[7];
    const float* lng = (const float*)d_in[8];
    const float* lnb = (const float*)d_in[9];
    const float* W2  = (const float*)d_in[10];
    const float* as2 = (const float*)d_in[11];
    const float* ad2 = (const float*)d_in[12];
    const float* b2  = (const float*)d_in[13];
    const float* fcw = (const float*)d_in[14];
    const float* fcb = (const float*)d_in[15];
    float* out = (float*)d_out;

    const int N = out_size;
    const int E = in_sizes[1] / 2;
    const int Etot = E + N;

    char* base = (char*)d_ws;
    size_t off = 0;
    auto carve = [&](size_t bytes) -> void* {
        off = (off + 255) & ~(size_t)255;
        void* p = base + off;
        off += bytes;
        return p;
    };
    _Float16* h0  = (_Float16*)carve((size_t)N * 160 * 2);
    _Float16* h1  = (_Float16*)carve((size_t)N * 320 * 2);
    float*    hp  = (float*)carve((size_t)N * HC * 4);
    float*    agg = (float*)carve((size_t)N * HC * 4);
    float*    als = (float*)carve((size_t)N * 2 * 4);
    float*    ald = (float*)carve((size_t)N * 2 * 4);
    unsigned* mb  = (unsigned*)carve((size_t)N * 2 * 4);
    float*    den = (float*)carve((size_t)N * 2 * 4);
    _Float16* pW1 = (_Float16*)carve((size_t)NTILES * 5 * 512 * 2);
    _Float16* pW2 = (_Float16*)carve((size_t)NTILES * 10 * 512 * 2);

    const int BLK = 256;
    auto gb = [](long long n, int b) { return (unsigned)((n + b - 1) / b); };

    // pack weights into WMMA fragment order
    k_packW<<<gb(NTILES * 5 * 32, BLK), BLK, 0, stream>>>(W1, pW1, Dd, HC, 5, NTILES);
    k_packW<<<gb(NTILES * 10 * 32, BLK), BLK, 0, stream>>>(W2, pW2, HC, HC, 10, NTILES);

    // argmax + embedding gather
    k_argmax_embed<<<gb((long long)N * 32, BLK), BLK, 0, stream>>>(x, emb, h0, N);

    // ===== layer 1 =====
    k_gemm<<<gb(N, 16), BLK, 0, stream>>>(h0, 160, 5, pW1, hp, N, HC);
    k_attn<<<gb((long long)N * 32, BLK), BLK, 0, stream>>>(hp, as1, ad1, als, ald, N);
    k_fill_u32<<<gb((long long)N * 2, BLK), BLK, 0, stream>>>(mb, 0u, (size_t)N * 2);
    k_fill_f32<<<gb((long long)N * 2, BLK), BLK, 0, stream>>>(den, 0.f, (size_t)N * 2);
    k_edge_max<<<gb(Etot, BLK), BLK, 0, stream>>>(ei, E, N, als, ald, mb);
    k_edge_expsum<<<gb(Etot, BLK), BLK, 0, stream>>>(ei, E, N, als, ald, mb, den);
    k_fill_f32<<<gb((long long)N * HC, BLK), BLK, 0, stream>>>(agg, 0.f, (size_t)N * HC);
    k_edge_agg<<<gb((long long)Etot * 32, BLK), BLK, 0, stream>>>(ei, E, N, als, ald, mb, den, hp, agg);
    k_ln<<<gb((long long)N * 32, BLK), BLK, 0, stream>>>(agg, b1, lng, lnb, h1, N);

    // ===== layer 2 =====
    k_gemm<<<gb(N, 16), BLK, 0, stream>>>(h1, 320, 10, pW2, hp, N, HC);
    k_attn<<<gb((long long)N * 32, BLK), BLK, 0, stream>>>(hp, as2, ad2, als, ald, N);
    k_fill_u32<<<gb((long long)N * 2, BLK), BLK, 0, stream>>>(mb, 0u, (size_t)N * 2);
    k_fill_f32<<<gb((long long)N * 2, BLK), BLK, 0, stream>>>(den, 0.f, (size_t)N * 2);
    k_edge_max<<<gb(Etot, BLK), BLK, 0, stream>>>(ei, E, N, als, ald, mb);
    k_edge_expsum<<<gb(Etot, BLK), BLK, 0, stream>>>(ei, E, N, als, ald, mb, den);
    k_fill_f32<<<gb((long long)N * HC, BLK), BLK, 0, stream>>>(agg, 0.f, (size_t)N * HC);
    k_edge_agg<<<gb((long long)Etot * 32, BLK), BLK, 0, stream>>>(ei, E, N, als, ald, mb, den, hp, agg);

    // final FC
    k_final<<<gb((long long)N * 32, BLK), BLK, 0, stream>>>(agg, b2, fcw, fcb, out, N);
}
